// TypeAreaScaledDotProductAttention_34497177321779
// MI455X (gfx1250) — compile-verified, matrix operand order checked by reading, not run on hardware
//
#include <hip/hip_runtime.h>

// Problem constants: B=4, H=16, S=2048, D=64 (fp32 in, fp32 out tuple)
#define BHN   64          // B*H
#define SEQ   2048
#define DIM   64
#define BM    128         // q rows per workgroup
#define WM    16          // q rows per wave
#define NWAVE 8
#define CHUNK 32          // k columns processed per inner iteration
#define NCH   (SEQ / CHUNK)

#define QP 72             // LDS pitch (bf16) for Q tile
#define KP 72             // LDS pitch (bf16) for K chunk
#define VP 40             // LDS pitch (bf16) for transposed V chunk [DIM][VP]

typedef __attribute__((ext_vector_type(16))) __bf16 v16bf;
typedef __attribute__((ext_vector_type(8)))  float  v8f;
typedef int v4i __attribute__((vector_size(16)));   // matches builtin signature

#define GLOBAL_AS __attribute__((address_space(1)))
#define LDS_AS    __attribute__((address_space(3)))

#if defined(__has_builtin)
#  if __has_builtin(__builtin_amdgcn_global_load_async_to_lds_b128)
#    define HAVE_ASYNC_LOAD 1
#  else
#    define HAVE_ASYNC_LOAD 0
#  endif
#  if __has_builtin(__builtin_amdgcn_global_store_async_from_lds_b128)
#    define HAVE_ASYNC_STORE 1
#  else
#    define HAVE_ASYNC_STORE 0
#  endif
#else
#  define HAVE_ASYNC_LOAD 0
#  define HAVE_ASYNC_STORE 0
#endif

__device__ __forceinline__ void wait_async0() {
#if defined(__has_builtin) && __has_builtin(__builtin_amdgcn_s_wait_asynccnt)
  __builtin_amdgcn_s_wait_asynccnt(0);
#else
  asm volatile("s_wait_asynccnt 0" ::: "memory");
#endif
}

__device__ __forceinline__ v8f wmma_bf16(v16bf a, v16bf b, v8f c) {
  // D = A(16x32 bf16) x B(32x16 bf16) + C(16x16 f32)
  return __builtin_amdgcn_wmma_f32_16x16x32_bf16(false, a, false, b, (short)0, c,
                                                 false, false);
}

__launch_bounds__(256, 1)
__global__ void attn_fused(const float* __restrict__ qa, const float* __restrict__ ka,
                           const float* __restrict__ va, const float* __restrict__ qs,
                           const float* __restrict__ ks, const float* __restrict__ vs,
                           const int*   __restrict__ mk,
                           float* __restrict__ out, float* __restrict__ attn)
{
  __shared__ __bf16 Qs[BM * QP];                 // 18.0 KB  Q tile (q_a+q_s)
  __shared__ __bf16 Ks[CHUNK * KP];              //  4.5 KB  K chunk
  __shared__ __bf16 Vt[DIM * VP];                //  5.0 KB  V chunk, transposed [d][k]
  __shared__ __bf16 Ps[NWAVE * WM * CHUNK];      //  8.0 KB  probs bf16 (A-frag bounce)
  __shared__ __align__(16) float Pf[NWAVE * WM * CHUNK]; // 16 KB probs f32
  __shared__ __align__(16) int   Ms[BM * CHUNK]; // 16 KB mask tile (async-staged)

  const int tid  = threadIdx.x;
  const int w    = tid >> 5;
  const int lane = tid & 31;
  const int half = lane >> 4;
  const int ln   = lane & 15;
  const int bh   = blockIdx.y;
  const int bb   = bh >> 4;                      // batch index (mask has no head dim)
  const int q0   = blockIdx.x * BM;
  const int qw   = q0 + w * WM;                  // this wave's first q row

  const size_t tb = (size_t)bh * SEQ * DIM;      // q/k/v head base

  // ---------------- stage Q tile: (q_a + q_s) -> bf16 LDS ----------------
  #pragma unroll
  for (int it = 0; it < (BM * DIM) / (256 * 4); ++it) {   // 8 iters
    const int e = (it * 256 + tid) * 4;
    const int r = e >> 6, d = e & 63;
    const size_t g = tb + (size_t)(q0 + r) * DIM + d;
    const float4 a = *(const float4*)(qa + g);
    const float4 s = *(const float4*)(qs + g);
    __bf16* p = &Qs[r * QP + d];
    p[0] = (__bf16)(a.x + s.x);
    p[1] = (__bf16)(a.y + s.y);
    p[2] = (__bf16)(a.z + s.z);
    p[3] = (__bf16)(a.w + s.w);
  }
  __syncthreads();

  // A-fragments of Q (16x32 each; k in [0,32) and [32,64)), resident all kernel.
  v16bf aQ0, aQ1;
  #pragma unroll
  for (int i = 0; i < 16; ++i) {
    const int kk = ((i >> 3) << 4) + (half << 3) + (i & 7);
    aQ0[i] = Qs[(w * WM + ln) * QP + kk];
    aQ1[i] = Qs[(w * WM + ln) * QP + 32 + kk];
  }

  // ================= PASS A: row sums of exp(masked scores) =================
  float lsum[8];
  #pragma unroll
  for (int r = 0; r < 8; ++r) lsum[r] = 0.0f;

  for (int ch = 0; ch < NCH; ++ch) {
    const int kc = ch * CHUNK;
    __syncthreads();

    // Async-stage mask tile (raw copy -> LDS, tracked on ASYNCcnt).
    {
      const size_t mg = ((size_t)bb * SEQ + q0) * SEQ + kc;
      #pragma unroll
      for (int it = 0; it < 4; ++it) {
        const int idx = it * 256 + tid;          // 1024 x 16B segments
        const int r = idx >> 3, seg = idx & 7;
#if HAVE_ASYNC_LOAD
        __builtin_amdgcn_global_load_async_to_lds_b128(
            (GLOBAL_AS v4i*)(mk + mg + (size_t)r * SEQ + seg * 4),
            (LDS_AS v4i*)&Ms[r * CHUNK + seg * 4], 0, 0);
#else
        *(int4*)&Ms[r * CHUNK + seg * 4] =
            *(const int4*)(mk + mg + (size_t)r * SEQ + seg * 4);
#endif
      }
    }

    #pragma unroll
    for (int it = 0; it < (CHUNK * DIM) / (256 * 4); ++it) {  // 2 iters
      const int e = (it * 256 + tid) * 4;
      const int kr = e >> 6, d = e & 63;
      const size_t g = tb + (size_t)(kc + kr) * DIM + d;
      const float4 a = *(const float4*)(ka + g);
      const float4 s = *(const float4*)(ks + g);
      __bf16* p = &Ks[kr * KP + d];
      p[0] = (__bf16)(a.x + s.x);
      p[1] = (__bf16)(a.y + s.y);
      p[2] = (__bf16)(a.z + s.z);
      p[3] = (__bf16)(a.w + s.w);
    }
    if (ch + 1 < NCH && tid < 64) {               // prefetch next chunk (128B/lane)
      const size_t g = tb + (size_t)(kc + CHUNK) * DIM + tid * 32;
      __builtin_prefetch(ka + g, 0, 0);
      __builtin_prefetch(ks + g, 0, 0);
    }
#if HAVE_ASYNC_LOAD
    wait_async0();
#endif
    __syncthreads();

    #pragma unroll
    for (int c = 0; c < 2; ++c) {
      v16bf bK0, bK1;
      #pragma unroll
      for (int i = 0; i < 16; ++i) {
        const int kk = ((i >> 3) << 4) + (half << 3) + (i & 7);
        bK0[i] = Ks[(c * 16 + ln) * KP + kk];
        bK1[i] = Ks[(c * 16 + ln) * KP + 32 + kk];
      }
      v8f acc = {0.f, 0.f, 0.f, 0.f, 0.f, 0.f, 0.f, 0.f};
      acc = wmma_bf16(aQ0, bK0, acc);
      acc = wmma_bf16(aQ1, bK1, acc);

      #pragma unroll
      for (int r = 0; r < 8; ++r) {
        const int mv = Ms[(w * WM + half * 8 + r) * CHUNK + c * 16 + ln];
        const float sc = (mv != 0) ? acc[r] * 0.125f : -1.0e9f;
        lsum[r] += __expf(sc);       // exp(-1e9) underflows to exactly 0
      }
    }
  }

  // Cross-lane row-sum reduction: 16 lanes of each half hold distinct columns.
  float linv[8];
  #pragma unroll
  for (int r = 0; r < 8; ++r) {
    float v = lsum[r];
    v += __shfl_xor(v, 1, 32);
    v += __shfl_xor(v, 2, 32);
    v += __shfl_xor(v, 4, 32);
    v += __shfl_xor(v, 8, 32);
    linv[r] = 1.0f / v;
  }

  // ====== PASS B: recompute scores, write normalized attn, out += P @ V ======
  v8f oacc[4];
  #pragma unroll
  for (int t = 0; t < 4; ++t)
    oacc[t] = (v8f){0.f, 0.f, 0.f, 0.f, 0.f, 0.f, 0.f, 0.f};

  for (int ch = 0; ch < NCH; ++ch) {
    const int kc = ch * CHUNK;
    __syncthreads();

    {
      const size_t mg = ((size_t)bb * SEQ + q0) * SEQ + kc;
      #pragma unroll
      for (int it = 0; it < 4; ++it) {
        const int idx = it * 256 + tid;
        const int r = idx >> 3, seg = idx & 7;
#if HAVE_ASYNC_LOAD
        __builtin_amdgcn_global_load_async_to_lds_b128(
            (GLOBAL_AS v4i*)(mk + mg + (size_t)r * SEQ + seg * 4),
            (LDS_AS v4i*)&Ms[r * CHUNK + seg * 4], 0, 0);
#else
        *(int4*)&Ms[r * CHUNK + seg * 4] =
            *(const int4*)(mk + mg + (size_t)r * SEQ + seg * 4);
#endif
      }
    }

    #pragma unroll
    for (int it = 0; it < (CHUNK * DIM) / (256 * 4); ++it) {
      const int e = (it * 256 + tid) * 4;
      const int kr = e >> 6, d = e & 63;
      const size_t g = tb + (size_t)(kc + kr) * DIM + d;
      const float4 a  = *(const float4*)(ka + g);
      const float4 s  = *(const float4*)(ks + g);
      __bf16* p = &Ks[kr * KP + d];
      p[0] = (__bf16)(a.x + s.x);
      p[1] = (__bf16)(a.y + s.y);
      p[2] = (__bf16)(a.z + s.z);
      p[3] = (__bf16)(a.w + s.w);
      const float4 a2 = *(const float4*)(va + g);
      const float4 s2 = *(const float4*)(vs + g);
      Vt[(d + 0) * VP + kr] = (__bf16)(a2.x + s2.x);  // transposed: [d][k]
      Vt[(d + 1) * VP + kr] = (__bf16)(a2.y + s2.y);
      Vt[(d + 2) * VP + kr] = (__bf16)(a2.z + s2.z);
      Vt[(d + 3) * VP + kr] = (__bf16)(a2.w + s2.w);
    }
    if (ch + 1 < NCH && tid < 64) {
      const size_t g = tb + (size_t)(kc + CHUNK) * DIM + tid * 32;
      __builtin_prefetch(ka + g, 0, 0);
      __builtin_prefetch(ks + g, 0, 0);
      __builtin_prefetch(va + g, 0, 0);
      __builtin_prefetch(vs + g, 0, 0);
    }
    // Drain mask async loads AND last chunk's async attn stores (Pf reuse).
#if HAVE_ASYNC_LOAD || HAVE_ASYNC_STORE
    wait_async0();
#endif
    __syncthreads();

    #pragma unroll
    for (int c = 0; c < 2; ++c) {
      v16bf bK0, bK1;
      #pragma unroll
      for (int i = 0; i < 16; ++i) {
        const int kk = ((i >> 3) << 4) + (half << 3) + (i & 7);
        bK0[i] = Ks[(c * 16 + ln) * KP + kk];
        bK1[i] = Ks[(c * 16 + ln) * KP + 32 + kk];
      }
      v8f acc = {0.f, 0.f, 0.f, 0.f, 0.f, 0.f, 0.f, 0.f};
      acc = wmma_bf16(aQ0, bK0, acc);
      acc = wmma_bf16(aQ1, bK1, acc);

      #pragma unroll
      for (int r = 0; r < 8; ++r) {
        const int mv = Ms[(w * WM + half * 8 + r) * CHUNK + c * 16 + ln];
        const float sc = (mv != 0) ? acc[r] * 0.125f : -1.0e9f;
        const float p = __expf(sc) * linv[r];
        const int idx = (w * WM + half * 8 + r) * CHUNK + c * 16 + ln;
        Ps[idx] = (__bf16)p;   // A-fragment re-layout bounce
        Pf[idx] = p;           // f32 copy for attn store
      }
    }

    // Wave-private LDS bounce: our ds stores must land before cross-lane reads
    // and before the async DMA reads Pf.
    asm volatile("s_wait_dscnt 0" ::: "memory");

    v16bf aP;
    #pragma unroll
    for (int i = 0; i < 16; ++i) {
      const int kk = ((i >> 3) << 4) + (half << 3) + (i & 7);
      aP[i] = Ps[(w * WM + ln) * CHUNK + kk];
    }
    #pragma unroll
    for (int t = 0; t < 4; ++t) {
      v16bf bV;
      #pragma unroll
      for (int i = 0; i < 16; ++i) {
        const int kk = ((i >> 3) << 4) + (half << 3) + (i & 7);
        bV[i] = Vt[(t * 16 + ln) * VP + kk];
      }
      oacc[t] = wmma_bf16(aP, bV, oacc[t]);
    }

    // Dominant 1.07 GB attn stream: ship straight from LDS, 16B per lane-op.
    #pragma unroll
    for (int g4 = 0; g4 < 4; ++g4) {
      const int flat = g4 * 128 + lane * 4;
      const int row = flat >> 5, col = flat & 31;
      const size_t adst = ((size_t)bh * SEQ + (qw + row)) * SEQ + kc + col;
#if HAVE_ASYNC_STORE
      __builtin_amdgcn_global_store_async_from_lds_b128(
          (GLOBAL_AS v4i*)(attn + adst),
          (LDS_AS v4i*)&Pf[w * WM * CHUNK + flat], 0, 0);
#else
      *(float4*)(attn + adst) = *(const float4*)&Pf[w * WM * CHUNK + flat];
#endif
    }
  }

#if HAVE_ASYNC_STORE
  wait_async0();                     // drain final attn stores
#endif

  // ---------------- store out (16x64 per wave, C-layout) ----------------
  #pragma unroll
  for (int t = 0; t < 4; ++t) {
    #pragma unroll
    for (int r = 0; r < 8; ++r) {
      out[((size_t)bh * SEQ + (qw + half * 8 + r)) * DIM + t * 16 + ln] = oacc[t][r];
    }
  }
}

extern "C" void kernel_launch(void* const* d_in, const int* in_sizes, int n_in,
                              void* d_out, int out_size, void* d_ws, size_t ws_size,
                              hipStream_t stream) {
  (void)in_sizes; (void)n_in; (void)d_ws; (void)ws_size; (void)out_size;
  const float* qa = (const float*)d_in[0];
  const float* ka = (const float*)d_in[1];
  const float* va = (const float*)d_in[2];
  const float* qs = (const float*)d_in[3];
  const float* ks = (const float*)d_in[4];
  const float* vs = (const float*)d_in[5];
  const int*   mk = (const int*)d_in[6];

  float* out  = (float*)d_out;
  float* attn = out + (size_t)BHN * SEQ * DIM;   // tuple: out first, then attn

  dim3 grid(SEQ / BM, BHN);                      // 16 q-tiles x 64 (b,h) heads
  attn_fused<<<grid, dim3(256), 0, stream>>>(qa, ka, va, qs, ks, vs, mk, out, attn);
}
